// GraphNetwork_56032143344052
// MI455X (gfx1250) — matrix operand order repaired and checked by reading, not compiled, not run on hardware
//
#include <hip/hip_runtime.h>

// ---------------------------------------------------------------------------
// CDNA5 (gfx1250) fused graph-network forward.
// All GEMMs run on v_wmma_f32_16x16x32_bf16 (wave32, 8 waves / 256-thread WG).
// A-tiles are staged to LDS with global_load_async_to_lds_b128 (ASYNCcnt),
// weight streams are helped along with unconditional global_prefetch_b8.
// ---------------------------------------------------------------------------

typedef __attribute__((ext_vector_type(16))) __bf16  v16bf;
typedef __attribute__((ext_vector_type(8)))  float   v8f;
typedef __attribute__((ext_vector_type(4)))  unsigned int u32x4;

union FragBf { v16bf v; u32x4 u[2]; };

#define BB   64     // batch
#define NN   49     // nodes (7x7)
#define NP   64     // padded nodes
#define FF   256    // hidden
#define CIN  1024   // conv1 input channels

__device__ __forceinline__ unsigned short f2bf(float f) {
  unsigned u = __float_as_uint(f);
  u += 0x7FFFu + ((u >> 16) & 1u);          // round-to-nearest-even
  return (unsigned short)(u >> 16);
}
__device__ __forceinline__ float bf2f(unsigned short h) {
  return __uint_as_float(((unsigned)h) << 16);
}

// ---------------------------------------------------------------------------
// elementwise fp32 -> bf16
// ---------------------------------------------------------------------------
__global__ void k_f32_to_bf16(const float* __restrict__ s, unsigned short* __restrict__ d, int n) {
  int i = blockIdx.x * 256 + threadIdx.x;
  if (i < n) d[i] = f2bf(s[i]);
}

// x[B,1024,7,7] -> xT_bf16[B, 64(pad rows n), 1024]
__global__ void k_prep_xT(const float* __restrict__ x, unsigned short* __restrict__ xT) {
  int idx = blockIdx.x * 256 + threadIdx.x;           // B*NP*CIN threads
  int c = idx & (CIN - 1);
  int n = (idx >> 10) & (NP - 1);
  int b = idx >> 16;
  float v = (n < NN) ? x[((long)b * CIN + c) * NN + n] : 0.f;
  xT[idx] = f2bf(v);
}

// edge init: ch0 = 0.5 + 0.5*eye, ch1 = 0.5. layout [B,2,49,49]
__global__ void k_init_edge(float* __restrict__ edge) {
  int idx = blockIdx.x * 256 + threadIdx.x;
  if (idx >= BB * 2 * NN * NN) return;
  int r = idx % (NN * NN);
  int c = (idx / (NN * NN)) & 1;
  int i = r / NN, j = r % NN;
  edge[idx] = (c == 0 && i == j) ? 1.0f : 0.5f;
}

// e = l1norm(edge * (1-eye)) along j, padded to [B*2, 64, 64] bf16
__global__ void k_enorm(const float* __restrict__ edge, unsigned short* __restrict__ enorm) {
  int i  = blockIdx.x & (NP - 1);
  int bc = blockIdx.x >> 6;          // b*2+c
  int j  = threadIdx.x;              // 64 threads
  __shared__ float red[64];
  float e = 0.f;
  if (i < NN && j < NN && j != i) e = edge[((long)bc * NN + i) * NN + j];
  red[j] = fabsf(e);
  __syncthreads();
  for (int off = 32; off >= 1; off >>= 1) {
    if (j < off) red[j] += red[j + off];
    __syncthreads();
  }
  float denom = fmaxf(red[0], 1e-12f);
  enorm[((long)bc * NP + i) * NP + j] = f2bf(e / denom);
}

// nodeT_bf[b, f, j] = node_f32[b, j, f]   (j padded to 64, pads are zero)
__global__ void k_nodeT(const float* __restrict__ node, unsigned short* __restrict__ nodeT) {
  int idx = blockIdx.x * 256 + threadIdx.x;           // B*FF*NP
  int j = idx & (NP - 1);
  int f = (idx >> 6) & (FF - 1);
  int b = idx >> 14;
  nodeT[idx] = f2bf(node[((long)b * NP + j) * FF + f]);
}

// hcat[b, m, 0:256] = node_bf[b, m, :]
__global__ void k_copy_node_h(const unsigned short* __restrict__ nodeb, unsigned short* __restrict__ hcat) {
  int idx = blockIdx.x * 256 + threadIdx.x;           // B*NP*FF
  int f = idx & (FF - 1);
  int m = (idx >> 8) & (NP - 1);
  int b = idx >> 14;
  hcat[((long)b * NP + m) * 768 + f] = nodeb[idx];
}

// ---------------------------------------------------------------------------
// Generic 64-row-block GEMM:  out = epilogue( A[g][64,K] @ W[Nout,K]^T )
// A is bf16 async-DMA'd into LDS; W is bf16 streamed from global (L2-resident).
// ---------------------------------------------------------------------------
__global__ __launch_bounds__(256) void k_gemm64(
    const unsigned short* __restrict__ A, const unsigned short* __restrict__ W,
    long wBatchStride, int bdiv, int K, int Nout,
    const float* __restrict__ bn_g, const float* __restrict__ bn_b,
    const float* __restrict__ bn_m, const float* __restrict__ bn_v,
    int do_lrelu,
    unsigned short* __restrict__ outBf, float* __restrict__ outF32,
    int outStride, int colOff0, int colOffStep, int rowsValid) {
  extern __shared__ unsigned short sA[];
  int g = blockIdx.x;
  int bg = g / bdiv, sub = g % bdiv;
  const unsigned short* Ag = A + (long)g * NP * K;
  const unsigned short* Wg = W + (long)bg * wBatchStride;

  { // stage A tile into LDS with CDNA5 async cache->LDS DMA (ASYNCcnt-tracked)
    int tot = NP * K / 8;                                   // 16B chunks
    unsigned ldsbase = (unsigned)(unsigned long long)(void*)sA;  // low 32b = LDS offset
    unsigned long long gbase = (unsigned long long)(const void*)Ag;
    for (int idx = threadIdx.x; idx < tot; idx += 256) {
      unsigned dst = ldsbase + (unsigned)idx * 16u;
      unsigned long long ga = gbase + (unsigned long long)idx * 16ull;
      asm volatile("global_load_async_to_lds_b128 %0, %1, off"
                   :: "v"(dst), "v"(ga) : "memory");
    }
    asm volatile("s_wait_asynccnt 0x0" ::: "memory");
  }
  __syncthreads();

  int lane = threadIdx.x & 31, wave = threadIdx.x >> 5;
  int ksel = lane >> 4, l15 = lane & 15;
  int numTiles = (Nout >> 4) * 4;
  int colOff = colOff0 + sub * colOffStep;

  for (int t = wave; t < numTiles; t += 8) {     // wave-uniform -> EXEC all 1s
    int mt = t & 3, nt = t >> 2;
    v8f acc = {};
    const unsigned short* arow = sA + (mt * 16 + l15) * K + 8 * ksel;
    const unsigned short* wrow = Wg + (long)(nt * 16 + l15) * K + 8 * ksel;
    for (int k0 = 0; k0 < K; k0 += 32) {
      // speculative prefetch four k-steps ahead (unconditional: overshoot is
      // harmless and keeps the inner loop branch-free / clause-friendly)
      __builtin_prefetch((const void*)(wrow + k0 + 128), 0, 3);
      FragBf a, w;
      a.u[0] = *(const u32x4*)(arow + k0);
      a.u[1] = *(const u32x4*)(arow + k0 + 16);
      w.u[0] = *(const u32x4*)(wrow + k0);
      w.u[1] = *(const u32x4*)(wrow + k0 + 16);
      acc = __builtin_amdgcn_wmma_f32_16x16x32_bf16(
          false, a.v, false, w.v, (short)0, acc, false, false);
    }
    int col = nt * 16 + l15;
    float gg = 1.f, be = 0.f, mm = 0.f, inv = 1.f;
    if (bn_g) { gg = bn_g[col]; be = bn_b[col]; mm = bn_m[col]; inv = rsqrtf(bn_v[col] + 1e-5f); }
#pragma unroll
    for (int r = 0; r < 8; r++) {
      int mrow = mt * 16 + r + 8 * ksel;
      float val = acc[r];
      if (bn_g) val = gg * (val - mm) * inv + be;
      if (do_lrelu) val = (val >= 0.f) ? val : 0.01f * val;
      if (mrow >= rowsValid) val = 0.f;
      long o = ((long)bg * NP + mrow) * (long)outStride + colOff + col;
      if (outF32) outF32[o] = val;
      if (outBf)  outBf[o] = f2bf(val);
    }
  }
}

// ---------------------------------------------------------------------------
// Fused edge MLP: per (b,i), run 64 j-rows through 4 GEMM+BN+LReLU stages
// entirely in LDS (ping-pong 2x 64KB), then sigmoid(h4 . wout + bout) -> sim.
// ---------------------------------------------------------------------------
struct EdgeArgs {
  const float* node;                 // [B,64,256] f32 (padded rows zero)
  const unsigned short* W[4];        // bf16 weights [Nout,K]
  const float* bn[4][4];             // g,b,m,v per stage
  const float* wout;                 // [256] f32
  const float* bout;                 // [1]  f32
  float* sim;                        // [B,49,49] f32
};

__device__ __forceinline__ void gemm_lds(
    const unsigned short* sA, int K,
    const unsigned short* __restrict__ W, int Nout,
    const float* const bnp[4], unsigned short* sOut) {
  int lane = threadIdx.x & 31, wave = threadIdx.x >> 5;
  int ksel = lane >> 4, l15 = lane & 15;
  int numTiles = (Nout >> 4) * 4;
  for (int t = wave; t < numTiles; t += 8) {
    int mt = t & 3, nt = t >> 2;
    v8f acc = {};
    const unsigned short* arow = sA + (mt * 16 + l15) * K + 8 * ksel;
    const unsigned short* wrow = W + (long)(nt * 16 + l15) * K + 8 * ksel;
    for (int k0 = 0; k0 < K; k0 += 32) {
      __builtin_prefetch((const void*)(wrow + k0 + 128), 0, 3);
      FragBf a, w;
      a.u[0] = *(const u32x4*)(arow + k0);
      a.u[1] = *(const u32x4*)(arow + k0 + 16);
      w.u[0] = *(const u32x4*)(wrow + k0);
      w.u[1] = *(const u32x4*)(wrow + k0 + 16);
      acc = __builtin_amdgcn_wmma_f32_16x16x32_bf16(
          false, a.v, false, w.v, (short)0, acc, false, false);
    }
    int col = nt * 16 + l15;
    float gg = bnp[0][col], be = bnp[1][col], mm = bnp[2][col];
    float inv = rsqrtf(bnp[3][col] + 1e-5f);
#pragma unroll
    for (int r = 0; r < 8; r++) {
      int mrow = mt * 16 + r + 8 * ksel;
      float val = gg * (acc[r] - mm) * inv + be;
      val = (val >= 0.f) ? val : 0.01f * val;
      sOut[mrow * Nout + col] = f2bf(val);
    }
  }
  __syncthreads();
}

__global__ __launch_bounds__(256) void k_edge_mlp(EdgeArgs ea) {
  extern __shared__ unsigned short sh[];
  unsigned short* s0 = sh;
  unsigned short* s1 = sh + NP * 512;
  int b = blockIdx.x / NN, i = blockIdx.x % NN;
  const float* nb = ea.node + (long)b * NP * FF;
  const float* ni = nb + (long)i * FF;

  // stage d[j,f] = |node_i - node_j|  (pad rows -> 0)
  for (int idx = threadIdx.x; idx < NP * FF; idx += 256) {
    int j = idx >> 8, f = idx & (FF - 1);
    float v = (j < NN) ? fabsf(ni[f] - nb[j * FF + f]) : 0.f;
    s0[idx] = f2bf(v);
  }
  __syncthreads();

  gemm_lds(s0, 256, ea.W[0], 512, ea.bn[0], s1);
  gemm_lds(s1, 512, ea.W[1], 512, ea.bn[1], s0);
  gemm_lds(s0, 512, ea.W[2], 256, ea.bn[2], s1);
  gemm_lds(s1, 256, ea.W[3], 256, ea.bn[3], s0);

  // sim[j] = sigmoid(h4[j] . wout + bout); each wave handles rows j = wave, wave+8, ...
  int lane = threadIdx.x & 31, wave = threadIdx.x >> 5;
  float bo = ea.bout[0];
  for (int j = wave; j < NN; j += 8) {
    const unsigned short* hr = s0 + j * 256 + lane * 8;
    float sum = 0.f;
#pragma unroll
    for (int e = 0; e < 8; e++) sum += bf2f(hr[e]) * ea.wout[lane * 8 + e];
#pragma unroll
    for (int off = 16; off >= 1; off >>= 1) sum += __shfl_xor(sum, off, 32);
    if (lane == 0)
      ea.sim[((long)b * NN + i) * NN + j] = 1.f / (1.f + __expf(-(sum + bo)));
  }
}

// ---------------------------------------------------------------------------
// Edge renormalization (per (b,i) row, in-place on edge)
// ---------------------------------------------------------------------------
__global__ void k_edge_fin(float* __restrict__ edge, const float* __restrict__ sim) {
  int b = blockIdx.x / NN, i = blockIdx.x % NN;
  int j = threadIdx.x;               // 64 threads
  __shared__ float red[4][64];
  float e0 = 0.f, e1 = 0.f, s = 0.f;
  if (j < NN) {
    float msk = (j == i) ? 0.f : 1.f;
    e0 = edge[(((long)b * 2 + 0) * NN + i) * NN + j] * msk;
    e1 = edge[(((long)b * 2 + 1) * NN + i) * NN + j] * msk;
    s  = sim[((long)b * NN + i) * NN + j];
  }
  float t0 = s * e0, t1 = (1.f - s) * e1;
  red[0][j] = e0; red[1][j] = e1; red[2][j] = fabsf(t0); red[3][j] = fabsf(t1);
  __syncthreads();
  for (int off = 32; off >= 1; off >>= 1) {
    if (j < off) {
#pragma unroll
      for (int c = 0; c < 4; c++) red[c][j] += red[c][j + off];
    }
    __syncthreads();
  }
  float se0 = red[0][0], se1 = red[1][0], st0 = red[2][0], st1 = red[3][0];
  if (j < NN) {
    float n0 = t0 / fmaxf(st0, 1e-12f) * se0;
    float n1 = t1 / fmaxf(st1, 1e-12f) * se1;
    float v0 = n0 + ((i == j) ? 1.f : 0.f) + 1e-6f;
    float v1 = n1 + 1e-6f;
    float den = v0 + v1;
    edge[(((long)b * 2 + 0) * NN + i) * NN + j] = v0 / den;
    edge[(((long)b * 2 + 1) * NN + i) * NN + j] = v1 / den;
  }
}

// feat[b,c] = mean_t node.flat[b][c*49+t]   (reproduces reshape(B,256,49) quirk)
__global__ void k_feat(const float* __restrict__ node, float* __restrict__ feat) {
  int b = blockIdx.x, c = threadIdx.x;       // 256 threads
  float sacc = 0.f;
  for (int t = 0; t < NN; t++) {
    int idx = c * NN + t;
    int n = idx >> 8, f = idx & (FF - 1);
    sacc += node[((long)b * NP + n) * FF + f];
  }
  feat[b * FF + c] = sacc * (1.f / 49.f);
}

// y = BN(feat @ last_w^T + last_b)
__global__ void k_final(const float* __restrict__ feat,
                        const float* __restrict__ lw, const float* __restrict__ lb,
                        const float* __restrict__ g, const float* __restrict__ be,
                        const float* __restrict__ m, const float* __restrict__ v,
                        float* __restrict__ out) {
  int idx = blockIdx.x * 256 + threadIdx.x;
  if (idx >= BB * 10) return;
  int b = idx / 10, o = idx % 10;
  float s = lb[o];
  for (int k = 0; k < FF; k++) s += feat[b * FF + k] * lw[o * FF + k];
  out[idx] = g[o] * (s - m[o]) * rsqrtf(v[o] + 1e-5f) + be[o];
}

// ---------------------------------------------------------------------------
// Host orchestration
// ---------------------------------------------------------------------------
extern "C" void kernel_launch(void* const* d_in, const int* in_sizes, int n_in,
                              void* d_out, int out_size, void* d_ws, size_t ws_size,
                              hipStream_t stream) {
  (void)in_sizes; (void)n_in; (void)out_size; (void)ws_size;
  auto fin = [&](int i) { return (const float*)d_in[i]; };

  size_t off = 0;
  auto alloc = [&](size_t bytes) -> void* {
    void* p = (char*)d_ws + off;
    off += (bytes + 255) & ~(size_t)255;
    return p;
  };

  unsigned short* xT    = (unsigned short*)alloc((size_t)BB * NP * CIN * 2);
  unsigned short* wconv = (unsigned short*)alloc((size_t)256 * CIN * 2);
  const int nwE[2] = {512 * 768, 256 * 512};
  const int ewE[4] = {512 * 256, 512 * 512, 256 * 512, 256 * 256};
  unsigned short* nw[2][2];
  unsigned short* ew[2][4];
  for (int l = 0; l < 2; l++) {
    for (int k = 0; k < 2; k++) nw[l][k] = (unsigned short*)alloc((size_t)nwE[k] * 2);
    for (int k = 0; k < 4; k++) ew[l][k] = (unsigned short*)alloc((size_t)ewE[k] * 2);
  }
  float*          node_f = (float*)alloc((size_t)BB * NP * FF * 4);
  unsigned short* node_b = (unsigned short*)alloc((size_t)BB * NP * FF * 2);
  unsigned short* nodeT  = (unsigned short*)alloc((size_t)BB * FF * NP * 2);
  unsigned short* enorm  = (unsigned short*)alloc((size_t)BB * 2 * NP * NP * 2);
  unsigned short* hcat   = (unsigned short*)alloc((size_t)BB * NP * 768 * 2);
  unsigned short* h1     = (unsigned short*)alloc((size_t)BB * NP * 512 * 2);
  float* edge = (float*)alloc((size_t)BB * 2 * NN * NN * 4);
  float* sim  = (float*)alloc((size_t)BB * NN * NN * 4);
  float* feat = (float*)alloc((size_t)BB * FF * 4);

  auto cvt = [&](const float* s, unsigned short* d, int n) {
    k_f32_to_bf16<<<(n + 255) / 256, 256, 0, stream>>>(s, d, n);
  };

  // weight conversions (deterministic, redone every call)
  cvt(fin(1), wconv, 256 * CIN);
  for (int l = 0; l < 2; l++) {
    int nb = 2 + 10 * l;
    cvt(fin(nb + 0), nw[l][0], nwE[0]);
    cvt(fin(nb + 1), nw[l][1], nwE[1]);
    int eb = 22 + 22 * l;
    for (int k = 0; k < 4; k++) cvt(fin(eb + k), ew[l][k], ewE[k]);
  }

  // conv1 (1x1) as GEMM
  k_prep_xT<<<(BB * NP * CIN) / 256, 256, 0, stream>>>(fin(0), xT);
  k_gemm64<<<BB, 256, NP * CIN * 2, stream>>>(
      xT, wconv, 0, 1, CIN, 256,
      nullptr, nullptr, nullptr, nullptr, 0,
      node_b, node_f, 256, 0, 0, NN);
  k_init_edge<<<(BB * 2 * NN * NN + 255) / 256, 256, 0, stream>>>(edge);

  for (int l = 0; l < 2; l++) {
    int nb = 2 + 10 * l, eb = 22 + 22 * l;

    // ---- node update ----
    k_enorm<<<BB * 2 * NP, 64, 0, stream>>>(edge, enorm);
    k_nodeT<<<(BB * FF * NP) / 256, 256, 0, stream>>>(node_f, nodeT);
    // agg[b,c] = enorm[b,c] @ node[b]  -> hcat cols 256/512
    k_gemm64<<<BB * 2, 256, NP * NP * 2, stream>>>(
        enorm, nodeT, (long)FF * NP, 2, NP, FF,
        nullptr, nullptr, nullptr, nullptr, 0,
        hcat, nullptr, 768, 256, 256, NN);
    k_copy_node_h<<<(BB * NP * FF) / 256, 256, 0, stream>>>(node_b, hcat);
    // node MLP layer 1: 768 -> 512
    k_gemm64<<<BB, 256, NP * 768 * 2, stream>>>(
        hcat, nw[l][0], 0, 1, 768, 512,
        fin(nb + 2), fin(nb + 3), fin(nb + 4), fin(nb + 5), 1,
        h1, nullptr, 512, 0, 0, NN);
    // node MLP layer 2: 512 -> 256
    k_gemm64<<<BB, 256, NP * 512 * 2, stream>>>(
        h1, nw[l][1], 0, 1, 512, 256,
        fin(nb + 6), fin(nb + 7), fin(nb + 8), fin(nb + 9), 1,
        node_b, node_f, 256, 0, 0, NN);

    // ---- edge update ----
    EdgeArgs ea;
    ea.node = node_f;
    for (int k = 0; k < 4; k++) {
      ea.W[k] = ew[l][k];
      for (int q = 0; q < 4; q++) ea.bn[k][q] = fin(eb + 4 + 4 * k + q);
    }
    ea.wout = fin(eb + 20);
    ea.bout = fin(eb + 21);
    ea.sim  = sim;
    k_edge_mlp<<<BB * NN, 256, 2 * NP * 512 * 2, stream>>>(ea);
    k_edge_fin<<<BB * NN, 64, 0, stream>>>(edge, sim);
  }

  k_feat<<<BB, 256, 0, stream>>>(node_f, feat);
  k_final<<<(BB * 10 + 255) / 256, 256, 0, stream>>>(
      feat, fin(66), fin(67), fin(68), fin(69), fin(70), fin(71), (float*)d_out);
}
